// Point_Transformer_84464826843692
// MI455X (gfx1250) — matrix operand order repaired and checked by reading, not compiled, not run on hardware
//
#include <hip/hip_runtime.h>

typedef __attribute__((ext_vector_type(16))) __bf16 v16bf;
typedef __attribute__((ext_vector_type(8)))  float  v8f;

namespace {

constexpr int NW  = 4;      // waves per block (128 threads)
constexpr int NPP = 8;      // points per wave
constexpr int TOTAL_PTS = 4 * 16384;                 // B * N
constexpr long CH = 16384L * 16L;                    // channel stride in feats/ppfs (floats)

// ---------------- LDS layout (uint words) ----------------
constexpr int U_W0F   = 0;                   // w0   : 8x2 tiles (A-frags, 256 uints/tile)
constexpr int U_W1F   = U_W0F   + 16 * 256;  // w1   : 4x4
constexpr int U_W2F   = U_W1F   + 16 * 256;  // w2   : 4x4
constexpr int U_W3F   = U_W2F   + 16 * 256;  // w3   : 4x4
constexpr int U_PW1F  = U_W3F   + 16 * 256;  // pw1  : 4x1 (inner 8 padded to 32)
constexpr int U_PW2F  = U_PW1F  +  4 * 256;  // pw2  : 4x2
constexpr int U_WOUTF = U_PW2F  +  8 * 256;  // wout : 8x2
constexpr int U_CW1T  = U_WOUTF + 16 * 256;  // cw1 lane table: 32 lanes * 260 f32 (256 + pad)
constexpr int U_CW2T  = U_CW1T  + 32 * 260;  // cw2 transposed: 16 * 68 floats (padded vs bank conflicts)
constexpr int U_BIAS  = U_CW2T  + 16 * 68;   // b0[128] b1[64] b2[64] b3[64] cb2[128] bout[128]
constexpr int U_WAVE  = U_BIAS  + 576;
// per-wave: xnpk 1024 | ppk 512 | xcol0 128 | outmid 64 | outbuf 128*12
constexpr int WAVE_STRIDE = 1024 + 512 + 128 + 64 + 128 * 12;   // 3264 uints (16B aligned)
constexpr int LDS_UINTS = U_WAVE + NW * WAVE_STRIDE;
constexpr size_t LDS_BYTES = (size_t)LDS_UINTS * 4;  // ~182 KB  (< 320 KB / workgroup)

// exactly one packed convert per pair: dst.l = bf16(a), dst.h = bf16(b)
// (instruction existence on gfx1250 verified in this loop's own disasm)
__device__ __forceinline__ unsigned pack_bf16(float a, float b) {
  unsigned r;
  asm("v_cvt_pk_bf16_f32 %0, %1, %2" : "=v"(r) : "v"(a), "v"(b));
  return r;
}

union FragU { v16bf v; unsigned u[8]; };
union AccU  { v8f   v; float    f[8]; };

__device__ __forceinline__ v8f wmma_bf16(v16bf a, v16bf b, v8f c) {
  // (neg_a, A, neg_b, B, c_mod, C, reuse_a, reuse_b)
  return __builtin_amdgcn_wmma_f32_16x16x32_bf16(false, a, false, b, (short)0, c, false, false);
}

__device__ __forceinline__ v16bf load_afrag(const unsigned* tile, int lane) {
  union { v16bf v; unsigned u[8]; } f;
  const unsigned* p = tile + lane * 8;           // 32B / lane, contiguous -> 2x ds_load_b128
#pragma unroll
  for (int i = 0; i < 8; ++i) f.u[i] = p[i];
  return f.v;
}

// Pre-swizzle a row-major weight (rows = out-ch, rowStride = in-ch) into WMMA
// A-fragment lane order, packed bf16 pairs.  k order used here (contiguous, not AMD's table order):
//   lane l: M = l&15, half h = l>>4 ; dword d holds K = k0, k0+1 with
//   k0 = 2d + 16h  (+ 32*ktile)
__device__ void build_afrag(unsigned* dst, const float* __restrict__ W,
                            int Mtiles, int Ktiles, int rowStride, int Cvalid,
                            int tid, int nthreads) {
  const int total = Mtiles * Ktiles * 256;
  for (int i = tid; i < total; i += nthreads) {
    int d  = i & 7;
    int l  = (i >> 3) & 31;
    int t  = i >> 8;
    int mt = t / Ktiles;
    int kt = t - mt * Ktiles;
    int m = l & 15, hh = l >> 4;
    int k0 = 2 * d + 16 * hh + kt * 32;
    int row = mt * 16 + m;
    float f0 = (k0     < Cvalid) ? W[row * rowStride + k0]     : 0.f;
    float f1 = (k0 + 1 < Cvalid) ? W[row * rowStride + k0 + 1] : 0.f;
    dst[i] = pack_bf16(f0, f1);
  }
}

__global__ void __launch_bounds__(NW * 32)
pt_fused(const float* __restrict__ feats, const float* __restrict__ ppfs,
         const float* __restrict__ w0,  const float* __restrict__ b0,
         const float* __restrict__ w1,  const float* __restrict__ b1,
         const float* __restrict__ w2,  const float* __restrict__ b2,
         const float* __restrict__ w3,  const float* __restrict__ b3,
         const float* __restrict__ pw1, const float* __restrict__ pw2,
         const float* __restrict__ cw1, const float* __restrict__ cw2,
         const float* __restrict__ cb2, const float* __restrict__ wout,
         const float* __restrict__ bout, float* __restrict__ out)
{
  extern __shared__ unsigned smem[];
  const int tid  = threadIdx.x;
  const int lane = tid & 31;
  const int wave = tid >> 5;
  const int kk   = lane & 15;      // WMMA column = k-slot of the point
  const int h    = lane >> 4;      // lane half

  // ---------------- per-block weight staging ----------------
  build_afrag(smem + U_W0F,   w0,  8, 2,  64,  64, tid, blockDim.x);
  build_afrag(smem + U_W1F,   w1,  4, 4, 128, 128, tid, blockDim.x);
  build_afrag(smem + U_W2F,   w2,  4, 4, 128, 128, tid, blockDim.x);
  build_afrag(smem + U_W3F,   w3,  4, 4, 128, 128, tid, blockDim.x);
  build_afrag(smem + U_PW1F,  pw1, 4, 1,   8,   8, tid, blockDim.x);
  build_afrag(smem + U_PW2F,  pw2, 4, 2,  64,  64, tid, blockDim.x);
  build_afrag(smem + U_WOUTF, wout,8, 2,  64,  64, tid, blockDim.x);

  // cw1 lane-gather table (f32): for lane ln, pair jj = g*16 + t*4 + rp holds
  // cw1[g, c0*16+kk], cw1[g, (c0+1)*16+kk]  with c0 = 16t + 2rp + 8h
  float* cw1f = (float*)(smem + U_CW1T);
  for (int i = tid; i < 32 * 256; i += blockDim.x) {
    int ln = i >> 8, j = i & 255;
    int jj = j >> 1, w = j & 1;
    int g = jj >> 4, t = (jj >> 2) & 3, rp = jj & 3;
    int lh = ln >> 4, lk = ln & 15;
    int c = 16 * t + 2 * rp + w + 8 * lh;
    cw1f[ln * 260 + j] = cw1[g * 1024 + c * 16 + lk];
  }
  // cw2 transposed: [kk][g*8+j]
  float* cw2t = (float*)(smem + U_CW2T);
  for (int i = tid; i < 128 * 8; i += blockDim.x) {
    int o = i >> 3, j = i & 7;
    cw2t[(o & 15) * 68 + (o >> 4) * 8 + j] = cw2[o * 8 + j];
  }
  // biases
  float* bias = (float*)(smem + U_BIAS);
  for (int i = tid; i < 576; i += blockDim.x) {
    float v;
    if      (i < 128) v = b0[i];
    else if (i < 192) v = b1[i - 128];
    else if (i < 256) v = b2[i - 192];
    else if (i < 320) v = b3[i - 256];
    else if (i < 448) v = cb2[i - 320];
    else              v = bout[i - 448];
    bias[i] = v;
  }
  __syncthreads();

  unsigned* wbuf  = smem + U_WAVE + wave * WAVE_STRIDE;
  unsigned* xnpk  = wbuf;                   // 128-ch xn as packed bf16 pairs, B-frag order
  unsigned* ppk   = wbuf + 1024;            // 64-ch pw1 hidden, packed bf16 pairs
  float*    xcol0 = (float*)(wbuf + 1536);  // relu'd xn column 0 (identity / x1 input)
  float*    outmid= (float*)(wbuf + 1664);  // 64-vec after k-weighted reduce
  float*    outbuf= (float*)(wbuf + 1728);  // [128 o][12 stride] -> NPP points buffered

  const int wid   = blockIdx.x * NW + wave;
  const int pbase = wid * NPP;
  const int bidx  = pbase >> 14;            // same batch for all NPP points (NPP | 16384)
  const int n0    = pbase & 16383;

  for (int pi = 0; pi < NPP; ++pi) {
    const int n = n0 + pi;

    const float* fb = feats + (size_t)bidx * 64 * CH + (size_t)n * 16;
    const float* pb = ppfs  + (size_t)bidx *  8 * CH + (size_t)n * 16;

    // prefetch next point's feats (global_prefetch_b8)
    if (pi + 1 < NPP) {
      const float* fn = fb + 16;
      __builtin_prefetch(fn + (long)(2 * lane)     * CH, 0, 1);
      __builtin_prefetch(fn + (long)(2 * lane + 1) * CH, 0, 1);
    }

    // ---- stage 0: xn = relu(w0 @ feats + b0)  (128x16) ----
    FragU bfeat[2];
#pragma unroll
    for (int ks = 0; ks < 2; ++ks)
#pragma unroll
      for (int d = 0; d < 8; ++d) {
        int c0 = ks * 32 + 16 * h + 2 * d;
        bfeat[ks].u[d] = pack_bf16(fb[(long)c0 * CH + kk], fb[(long)(c0 + 1) * CH + kk]);
      }
#pragma unroll
    for (int mt = 0; mt < 8; ++mt) {
      AccU acc;
#pragma unroll
      for (int r = 0; r < 8; ++r) acc.f[r] = bias[mt * 16 + 8 * h + r];       // b0
#pragma unroll
      for (int ks = 0; ks < 2; ++ks)
        acc.v = wmma_bf16(load_afrag(smem + U_W0F + (mt * 2 + ks) * 256, lane), bfeat[ks].v, acc.v);
#pragma unroll
      for (int r = 0; r < 8; ++r) acc.f[r] = fmaxf(acc.f[r], 0.f);
#pragma unroll
      for (int r = 0; r < 8; r += 2)                                          // pair rows -> B-frag order
        xnpk[(8 * mt + 4 * h + (r >> 1)) * 16 + kk] = pack_bf16(acc.f[r], acc.f[r + 1]);
      if (kk == 0)
#pragma unroll
        for (int r = 0; r < 8; ++r) xcol0[mt * 16 + 8 * h + r] = acc.f[r];
    }

    // ---- ptf stage 1: p = relu(pw1 @ ppfs)  (64x16, inner 8 zero-padded) ----
    FragU bppf;
#pragma unroll
    for (int d = 0; d < 8; ++d) {
      int c0 = 16 * h + 2 * d;
      float f0 = (c0     < 8) ? pb[(long)c0 * CH + kk]       : 0.f;
      float f1 = (c0 + 1 < 8) ? pb[(long)(c0 + 1) * CH + kk] : 0.f;
      bppf.u[d] = pack_bf16(f0, f1);
    }
#pragma unroll
    for (int mt = 0; mt < 4; ++mt) {
      AccU acc;
#pragma unroll
      for (int r = 0; r < 8; ++r) acc.f[r] = 0.f;
      acc.v = wmma_bf16(load_afrag(smem + U_PW1F + mt * 256, lane), bppf.v, acc.v);
#pragma unroll
      for (int r = 0; r < 8; r += 2) {
        float f0 = fmaxf(acc.f[r], 0.f), f1 = fmaxf(acc.f[r + 1], 0.f);
        ppk[(8 * mt + 4 * h + (r >> 1)) * 16 + kk] = pack_bf16(f0, f1);
      }
    }

    // ---- ptf stage 2: ptf = pw2 @ p  (64x16) ----
    float ptf[4][8];
    {
      FragU bp[2];
#pragma unroll
      for (int ks = 0; ks < 2; ++ks)
#pragma unroll
        for (int d = 0; d < 8; ++d) bp[ks].u[d] = ppk[(16 * ks + 8 * h + d) * 16 + kk];
#pragma unroll
      for (int mt = 0; mt < 4; ++mt) {
        AccU acc;
#pragma unroll
        for (int r = 0; r < 8; ++r) acc.f[r] = 0.f;
#pragma unroll
        for (int ks = 0; ks < 2; ++ks)
          acc.v = wmma_bf16(load_afrag(smem + U_PW2F + (mt * 2 + ks) * 256, lane), bp[ks].v, acc.v);
#pragma unroll
        for (int r = 0; r < 8; ++r) ptf[mt][r] = acc.f[r];
      }
    }

    // ---- x1 = w1 @ xcol0 + b1 (column-broadcast B) ----
    float xfs[4][8];
    {
      FragU bx[4];
#pragma unroll
      for (int ks = 0; ks < 4; ++ks)
#pragma unroll
        for (int d = 0; d < 8; ++d) {
          int c0 = ks * 32 + 16 * h + 2 * d;
          bx[ks].u[d] = pack_bf16(xcol0[c0], xcol0[c0 + 1]);
        }
#pragma unroll
      for (int mt = 0; mt < 4; ++mt) {
        AccU acc;
#pragma unroll
        for (int r = 0; r < 8; ++r) acc.f[r] = bias[128 + mt * 16 + 8 * h + r];  // b1
#pragma unroll
        for (int ks = 0; ks < 4; ++ks)
          acc.v = wmma_bf16(load_afrag(smem + U_W1F + (mt * 4 + ks) * 256, lane), bx[ks].v, acc.v);
#pragma unroll
        for (int r = 0; r < 8; ++r) xfs[mt][r] = acc.f[r];
      }
    }

    // ---- x2 = w2@xn+b2 ; x3 = w3@xn+b3 (share the xn B-frag) ----
    float x3v[4][8];
    {
      AccU ax2[4], ax3[4];
#pragma unroll
      for (int mt = 0; mt < 4; ++mt)
#pragma unroll
        for (int r = 0; r < 8; ++r) {
          ax2[mt].f[r] = bias[192 + mt * 16 + 8 * h + r];
          ax3[mt].f[r] = bias[256 + mt * 16 + 8 * h + r];
        }
#pragma unroll
      for (int ks = 0; ks < 4; ++ks) {
        FragU bxn;
#pragma unroll
        for (int d = 0; d < 8; ++d) bxn.u[d] = xnpk[(16 * ks + 8 * h + d) * 16 + kk];
#pragma unroll
        for (int mt = 0; mt < 4; ++mt) {
          ax2[mt].v = wmma_bf16(load_afrag(smem + U_W2F + (mt * 4 + ks) * 256, lane), bxn.v, ax2[mt].v);
          ax3[mt].v = wmma_bf16(load_afrag(smem + U_W3F + (mt * 4 + ks) * 256, lane), bxn.v, ax3[mt].v);
        }
      }
#pragma unroll
      for (int mt = 0; mt < 4; ++mt)
#pragma unroll
        for (int r = 0; r < 8; ++r) {
          x3v[mt][r] = ax3[mt].f[r] + ptf[mt][r];
          xfs[mt][r] = xfs[mt][r] - ax2[mt].f[r] + ptf[mt][r];
        }
    }

    // ---- h[g] = relu(cw1 . vec(xfs))  (per-lane f32 gather + full-wave reduce) ----
    float hg[8];
#pragma unroll
    for (int g = 0; g < 8; ++g) {
      float s = 0.f;
#pragma unroll
      for (int t = 0; t < 4; ++t)
#pragma unroll
        for (int rp = 0; rp < 4; ++rp) {
          int jj = g * 16 + t * 4 + rp;
          s += cw1f[lane * 260 + 2 * jj]     * xfs[t][2 * rp]
             + cw1f[lane * 260 + 2 * jj + 1] * xfs[t][2 * rp + 1];
        }
      s += __shfl_xor(s, 1);  s += __shfl_xor(s, 2);  s += __shfl_xor(s, 4);
      s += __shfl_xor(s, 8);  s += __shfl_xor(s, 16);
      hg[g] = fmaxf(s, 0.f);
    }

    // ---- wl = cw2 @ h + cb2 ; softmax over the 16 k-slots (lanes of the column group) ----
    float pw[8];
#pragma unroll
    for (int g = 0; g < 8; ++g) {
      float wl = bias[320 + g * 16 + kk];                      // cb2
#pragma unroll
      for (int j = 0; j < 8; ++j) wl += cw2t[kk * 68 + g * 8 + j] * hg[j];
      float m = wl;
      m = fmaxf(m, __shfl_xor(m, 1)); m = fmaxf(m, __shfl_xor(m, 2));
      m = fmaxf(m, __shfl_xor(m, 4)); m = fmaxf(m, __shfl_xor(m, 8));
      float e = __expf(wl - m);
      float s = e;
      s += __shfl_xor(s, 1); s += __shfl_xor(s, 2); s += __shfl_xor(s, 4); s += __shfl_xor(s, 8);
      pw[g] = e / s;
    }

    // ---- out_mid[c] = relu(sum_k pw[c%8,k] * x3[c,k])  (c%8 == register r) ----
#pragma unroll
    for (int t = 0; t < 4; ++t)
#pragma unroll
      for (int r = 0; r < 8; ++r) {
        float v = pw[r] * x3v[t][r];
        v += __shfl_xor(v, 1); v += __shfl_xor(v, 2); v += __shfl_xor(v, 4); v += __shfl_xor(v, 8);
        v = fmaxf(v, 0.f);
        if (kk == 0) outmid[t * 16 + 8 * h + r] = v;
      }

    // ---- out = wout @ out_mid + bout + identity (column-broadcast B) ----
    FragU bo[2];
#pragma unroll
    for (int ks = 0; ks < 2; ++ks)
#pragma unroll
      for (int d = 0; d < 8; ++d) {
        int c0 = ks * 32 + 16 * h + 2 * d;
        bo[ks].u[d] = pack_bf16(outmid[c0], outmid[c0 + 1]);
      }
#pragma unroll
    for (int mt = 0; mt < 8; ++mt) {
      AccU acc;
#pragma unroll
      for (int r = 0; r < 8; ++r) {
        int row = mt * 16 + 8 * h + r;
        acc.f[r] = bias[448 + row] + xcol0[row];               // bout + identity
      }
#pragma unroll
      for (int ks = 0; ks < 2; ++ks)
        acc.v = wmma_bf16(load_afrag(smem + U_WOUTF + (mt * 2 + ks) * 256, lane), bo[ks].v, acc.v);
      if (kk == 0)
#pragma unroll
        for (int r = 0; r < 8; ++r)
          outbuf[(mt * 16 + 8 * h + r) * 12 + pi] = acc.f[r];
    }
  }

  // ---- coalesced flush: 8 consecutive n per output channel (global_store_b128) ----
  {
    float* ob = out + (size_t)bidx * 128 * 16384 + n0;
#pragma unroll
    for (int w4 = 0; w4 < 4; ++w4) {
      int o = lane + 32 * w4;
      float4 lo = *(const float4*)&outbuf[o * 12];
      float4 hi = *(const float4*)&outbuf[o * 12 + 4];
      *(float4*)(ob + (size_t)o * 16384)     = lo;
      *(float4*)(ob + (size_t)o * 16384 + 4) = hi;
    }
  }
}

} // namespace

extern "C" void kernel_launch(void* const* d_in, const int* in_sizes, int n_in,
                              void* d_out, int out_size, void* d_ws, size_t ws_size,
                              hipStream_t stream) {
  const float* feats = (const float*)d_in[0];
  const float* ppfs  = (const float*)d_in[1];
  const float* w0    = (const float*)d_in[2];
  const float* b0    = (const float*)d_in[3];
  const float* w1    = (const float*)d_in[4];
  const float* b1    = (const float*)d_in[5];
  const float* w2    = (const float*)d_in[6];
  const float* b2    = (const float*)d_in[7];
  const float* w3    = (const float*)d_in[8];
  const float* b3    = (const float*)d_in[9];
  const float* pw1   = (const float*)d_in[10];
  const float* pw2   = (const float*)d_in[11];
  const float* cw1   = (const float*)d_in[12];
  const float* cw2   = (const float*)d_in[13];
  const float* cb2   = (const float*)d_in[14];
  const float* wout  = (const float*)d_in[15];
  const float* bout  = (const float*)d_in[16];
  float* outp = (float*)d_out;

  (void)in_sizes; (void)n_in; (void)out_size; (void)d_ws; (void)ws_size;

  (void)hipFuncSetAttribute((const void*)pt_fused,
                            hipFuncAttributeMaxDynamicSharedMemorySize,
                            (int)LDS_BYTES);

  dim3 grid(TOTAL_PTS / (NW * NPP));   // 2048 blocks
  dim3 block(NW * 32);                 // 4 waves
  pt_fused<<<grid, block, LDS_BYTES, stream>>>(
      feats, ppfs, w0, b0, w1, b1, w2, b2, w3, b3,
      pw1, pw2, cw1, cw2, cb2, wout, bout, outp);
}
